// HybridEncoder_87608742904434
// MI455X (gfx1250) — matrix operand order repaired and checked by reading, not compiled, not run on hardware
//
#include <hip/hip_runtime.h>
#include <hip/hip_bf16.h>

typedef __attribute__((ext_vector_type(16))) __bf16 v16bf;
typedef __attribute__((ext_vector_type(8)))  float  v8f;

#define NDIM 3
#define EDIM 2
#define NHEAD 4
#define ENC_NEG_INF 0x007FFFFFu

// ---- order-preserving float<->uint encoding for atomic max on f32 ----
__device__ __forceinline__ unsigned fenc(float f){
    unsigned u = __float_as_uint(f);
    return (u & 0x80000000u) ? ~u : (u | 0x80000000u);
}
__device__ __forceinline__ float fdec(unsigned e){
    unsigned u = (e & 0x80000000u) ? (e & 0x7FFFFFFFu) : ~e;
    return __uint_as_float(u);
}

// ---- f32 -> bf16 (round-nearest-even), stored as ushort ----
__device__ __forceinline__ unsigned short f2bfu(float f){
    unsigned x = __float_as_uint(f);
    return (unsigned short)((x + 0x7FFFu + ((x >> 16) & 1u)) >> 16);
}
__device__ __forceinline__ __bf16 us2bf(unsigned short s){
    union { unsigned short u; __bf16 b; } c; c.u = s; return c.b;
}

// ---- WMMA A-fragment load (wave32, contiguous k order, same as the B swizzle) ----
// A 16x32 bf16: lane L -> row m=L&15, half=L>>4; elem i -> K = k0 + (half<<4) + i
// (one contiguous 16-element run, matching the B swizzle)
__device__ __forceinline__ v16bf load_a_frag(const unsigned short* tile, int ld, int lane, int k0){
    int m = lane & 15, half = lane >> 4;
    v16bf a;
#pragma unroll
    for (int i = 0; i < 16; ++i){
        int k = k0 + (half << 4) + i;
        a[i] = us2bf(tile[m * ld + k]);
    }
    return a;
}
// B fragments come from pre-swizzled buffers: lane's 16 elements contiguous (32B)
__device__ __forceinline__ v16bf load_b_swz(const unsigned short* p){
    return *(const v16bf*)p;
}

// ================= K1: init scratch =================
__global__ void k_init(unsigned* xne, int n1, unsigned* menc, int n2, float* ssum, int n3){
    int i = blockIdx.x * blockDim.x + threadIdx.x;
    if (i < n1) xne[i]  = ENC_NEG_INF;
    if (i < n2) menc[i] = ENC_NEG_INF;
    if (i < n3) ssum[i] = 0.f;
}

// ================= K2a: W2 (64x64) -> bf16, B-fragment-swizzled =================
// layout p = (((kt*4 + nt)*32 + lane)*16 + ii); K = kt*32 + (lane>>4)*16 + ii; N = nt*16 + (lane&15)
__global__ void k_w2swz(const float* __restrict__ W2, unsigned short* __restrict__ w2s){
    int i = blockIdx.x * blockDim.x + threadIdx.x;
    if (i >= 4096) return;
    int ii = i & 15, lane = (i >> 4) & 31, nt = (i >> 9) & 3, kt = (i >> 11) & 1;
    int k = kt * 32 + (lane >> 4) * 16 + ii;
    int n = nt * 16 + (lane & 15);
    w2s[i] = f2bfu(W2[k * 64 + n]);
}

// ================= K2b: Wg (128x256) -> bf16, B-fragment-swizzled =================
// p = ((((nh*4 + kt)*8 + ntl)*32 + lane)*16 + ii); K = kt*32+(lane>>4)*16+ii; N = nh*128+ntl*16+(lane&15)
__global__ void k_wgswz(const float* __restrict__ Wg, unsigned short* __restrict__ wgs){
    int i = blockIdx.x * blockDim.x + threadIdx.x;
    if (i >= 32768) return;
    int ii = i & 15, lane = (i >> 4) & 31, ntl = (i >> 9) & 7, kt = (i >> 12) & 3, nh = (i >> 14) & 1;
    int k = kt * 32 + (lane >> 4) * 16 + ii;
    int n = nh * 128 + ntl * 16 + (lane & 15);
    wgs[i] = f2bfu(Wg[k * 256 + n]);
}

// ================= K3: EdgeConv MLP (WMMA) + scatter-max =================
// block = 128 threads (4 waves), 16 edges per wave. Tail handling is pure arithmetic:
// invalid rows get +(-inf) before encoding -> key == ENC_NEG_INF -> atomicMax no-op.
__global__ void k_edgeconv(const float* __restrict__ x,
                           const int* __restrict__ row, const int* __restrict__ col,
                           const float* __restrict__ W1, const float* __restrict__ b1,
                           const unsigned short* __restrict__ w2s, const float* __restrict__ b2,
                           unsigned* __restrict__ xnode_enc, int E){
    __shared__ __align__(32) unsigned short sW2s[4096];   // 8 KB, swizzled
    __shared__ __align__(32) unsigned short sA[4][16 * 64];
    __shared__ __align__(32) int   sCol[4][16];
    __shared__ float sW1[6 * 64];
    __shared__ float sB1[64], sB2[64];
    __shared__ float sMsg[4][16 * 6];

    int tid = threadIdx.x, w = tid >> 5, lane = tid & 31;
    {   // stage swizzled W2 (bulk b128 copy)
        const uint4* s = (const uint4*)w2s;
        uint4* d = (uint4*)sW2s;
        for (int i = tid; i < 512; i += 128) d[i] = s[i];
    }
    for (int i = tid; i < 384; i += 128) sW1[i] = W1[i];
    if (tid < 64){ sB1[tid] = b1[tid]; sB2[tid] = b2[tid]; }

    int eBase = blockIdx.x * 64 + w * 16;
    if (lane < 16){
        int e = eBase + lane;
        int ri = 0, ci = 0;
        if (e < E){ ri = row[e]; ci = col[e]; }
        sCol[w][lane] = ci;
#pragma unroll
        for (int d = 0; d < NDIM; ++d){
            float xi = (e < E) ? x[ci * NDIM + d] : 0.f;
            float xj = (e < E) ? x[ri * NDIM + d] : 0.f;
            sMsg[w][lane * 6 + d]     = xi;
            sMsg[w][lane * 6 + 3 + d] = xj - xi;
        }
    }
    __syncthreads();

    // h1 = relu(msg @ W1 + b1); each lane owns half a row, pair-packed bf16 stores
    {
        int m = lane >> 1, ob = (lane & 1) * 32;
        float mg[6];
#pragma unroll
        for (int d = 0; d < 6; ++d) mg[d] = sMsg[w][m * 6 + d];
        unsigned* dst = (unsigned*)(&sA[w][m * 64 + ob]);
#pragma unroll
        for (int k = 0; k < 32; k += 2){
            int o = ob + k;
            float s0 = sB1[o], s1 = sB1[o + 1];
#pragma unroll
            for (int d = 0; d < 6; ++d){
                s0 += mg[d] * sW1[d * 64 + o];
                s1 += mg[d] * sW1[d * 64 + o + 1];
            }
            s0 = s0 > 0.f ? s0 : 0.f;
            s1 = s1 > 0.f ? s1 : 0.f;
            dst[k >> 1] = (unsigned)f2bfu(s0) | ((unsigned)f2bfu(s1) << 16);
        }
    }

    v16bf a0 = load_a_frag(&sA[w][0], 64, lane, 0);
    v16bf a1 = load_a_frag(&sA[w][0], 64, lane, 32);
    int half = lane >> 4, nL = lane & 15;

    int4 c0 = *(const int4*)(&sCol[w][8 * half]);
    int4 c1 = *(const int4*)(&sCol[w][8 * half + 4]);
    int myCol[8] = {c0.x, c0.y, c0.z, c0.w, c1.x, c1.y, c1.z, c1.w};
    int eRow = eBase + 8 * half;   // first edge row this lane's accumulators cover

    // per-row invalidation addend: 0 for valid rows, -inf for tail rows (hoisted)
    float addend[8];
#pragma unroll
    for (int r = 0; r < 8; ++r)
        addend[r] = (eRow + r < E) ? 0.f : __uint_as_float(0xFF800000u);

#pragma unroll
    for (int nt = 0; nt < 4; ++nt){
        v8f c;
#pragma unroll
        for (int i = 0; i < 8; ++i) c[i] = 0.f;
        v16bf bf0 = load_b_swz(sW2s + ((0 * 4 + nt) * 32 + lane) * 16);
        v16bf bf1 = load_b_swz(sW2s + ((1 * 4 + nt) * 32 + lane) * 16);
        c = __builtin_amdgcn_wmma_f32_16x16x32_bf16(false, a0, false, bf0, (short)0, c, false, false);
        c = __builtin_amdgcn_wmma_f32_16x16x32_bf16(false, a1, false, bf1, (short)0, c, false, false);
        int n = nt * 16 + nL;
        float bias = sB2[n];
#pragma unroll
        for (int r = 0; r < 8; ++r){
            unsigned key = fenc(c[r] + bias + addend[r]);     // -inf -> ENC_NEG_INF -> no-op
            atomicMax(&xnode_enc[myCol[r] * 64 + n], key);
        }
    }
}

// ================= K4: decode node max -> bf16 (empty segments -> 0) =================
__global__ void k_decode(const unsigned* __restrict__ xne, unsigned short* __restrict__ xnb, int n){
    int i = blockIdx.x * blockDim.x + threadIdx.x;
    if (i >= n) return;
    unsigned e = xne[i];
    float v = (e == ENC_NEG_INF) ? 0.f : fdec(e);
    xnb[i] = f2bfu(v);
}

// ================= K5: xg = [x_node[row], x_edge] @ Wg (WMMA) =================
// block = 128 threads (4 waves), 16 edges per wave; swizzled Wg staged in two 32KB halves.
// xg is padded to a multiple of 64 rows -> stores are unconditional (tail rows never read).
__global__ void k_gat_gemm(const unsigned short* __restrict__ xnb, const int* __restrict__ row,
                           const float* __restrict__ ea, const float* __restrict__ We,
                           const float* __restrict__ be, const unsigned short* __restrict__ wgs,
                           float* __restrict__ xg, int E){
    __shared__ __align__(32) unsigned short sWg[16384];   // 32 KB, one n-half, swizzled
    __shared__ __align__(32) unsigned short sA[4][16 * 128];
    __shared__ int   sRow[4][16];
    __shared__ float sEa[4][16 * 2];

    int tid = threadIdx.x, w = tid >> 5, lane = tid & 31;
    int eBase = blockIdx.x * 64 + w * 16;

    if (lane < 16){
        int e = eBase + lane;
        sRow[w][lane]        = (e < E) ? row[e] : 0;
        sEa[w][lane * 2]     = (e < E) ? ea[e * 2]     : 0.f;
        sEa[w][lane * 2 + 1] = (e < E) ? ea[e * 2 + 1] : 0.f;
    }
    {   // each lane owns half a row of the A tile
        int m = lane >> 1, ob = (lane & 1) * 32;
        // cols 0..63: gather bf16 node features (b128 loads/stores)
        int r = sRow[w][m];
        const uint4* s = (const uint4*)(xnb + r * 64 + ob);
        uint4* d = (uint4*)(&sA[w][m * 128 + ob]);
#pragma unroll
        for (int k = 0; k < 4; ++k) d[k] = s[k];
        // cols 64..127: edge-feature MLP relu(edge_attr @ We + be), pair-packed stores
        float e0 = sEa[w][m * 2], e1 = sEa[w][m * 2 + 1];
        unsigned* d2 = (unsigned*)(&sA[w][m * 128 + 64 + ob]);
#pragma unroll
        for (int k = 0; k < 32; k += 2){
            int o = ob + k;
            float v0 = be[o]     + e0 * We[o]     + e1 * We[64 + o];
            float v1 = be[o + 1] + e0 * We[o + 1] + e1 * We[64 + o + 1];
            v0 = v0 > 0.f ? v0 : 0.f;
            v1 = v1 > 0.f ? v1 : 0.f;
            d2[k >> 1] = (unsigned)f2bfu(v0) | ((unsigned)f2bfu(v1) << 16);
        }
    }

    v16bf a0 = load_a_frag(&sA[w][0], 128, lane, 0);
    v16bf a1 = load_a_frag(&sA[w][0], 128, lane, 32);
    v16bf a2 = load_a_frag(&sA[w][0], 128, lane, 64);
    v16bf a3 = load_a_frag(&sA[w][0], 128, lane, 96);
    int half = lane >> 4, nL = lane & 15;

    for (int nh = 0; nh < 2; ++nh){
        __syncthreads();
        {
            const uint4* s = (const uint4*)(wgs + nh * 16384);
            uint4* d = (uint4*)sWg;
            for (int i = tid; i < 2048; i += 128) d[i] = s[i];
        }
        __syncthreads();
#pragma unroll
        for (int nt = 0; nt < 8; ++nt){
            v8f c;
#pragma unroll
            for (int i = 0; i < 8; ++i) c[i] = 0.f;
            c = __builtin_amdgcn_wmma_f32_16x16x32_bf16(false, a0, false, load_b_swz(sWg + ((0 * 8 + nt) * 32 + lane) * 16), (short)0, c, false, false);
            c = __builtin_amdgcn_wmma_f32_16x16x32_bf16(false, a1, false, load_b_swz(sWg + ((1 * 8 + nt) * 32 + lane) * 16), (short)0, c, false, false);
            c = __builtin_amdgcn_wmma_f32_16x16x32_bf16(false, a2, false, load_b_swz(sWg + ((2 * 8 + nt) * 32 + lane) * 16), (short)0, c, false, false);
            c = __builtin_amdgcn_wmma_f32_16x16x32_bf16(false, a3, false, load_b_swz(sWg + ((3 * 8 + nt) * 32 + lane) * 16), (short)0, c, false, false);
            int n = nh * 128 + nt * 16 + nL;
            float* o = &xg[(eBase + 8 * half) * 256 + n];
#pragma unroll
            for (int r = 0; r < 8; ++r) o[r * 256] = c[r];   // unconditional: xg padded
        }
    }
}

// ================= K6: attention coefficients (float4) =================
__global__ void k_att(const float* __restrict__ xg, const float* __restrict__ atts,
                      const float* __restrict__ attd, float* __restrict__ as_,
                      float* __restrict__ ad_, int E){
    int i = blockIdx.x * blockDim.x + threadIdx.x;
    if (i >= E * NHEAD) return;
    int e = i >> 2, h = i & 3;
    const float4* v  = (const float4*)(xg + e * 256 + h * 64);
    const float4* p1 = (const float4*)(atts + h * 64);
    const float4* p2 = (const float4*)(attd + h * 64);
    float s1 = 0.f, s2 = 0.f;
#pragma unroll
    for (int c = 0; c < 16; ++c){
        float4 t = v[c], a = p1[c], b = p2[c];
        s1 += t.x * a.x + t.y * a.y + t.z * a.z + t.w * a.w;
        s2 += t.x * b.x + t.y * b.y + t.z * b.z + t.w * b.w;
    }
    as_[i] = s1; ad_[i] = s2;
}

// ================= K7: logits + segment max =================
__global__ void k_logits(const float* __restrict__ as_, const float* __restrict__ ad_,
                         const int* __restrict__ row, const int* __restrict__ col,
                         float* __restrict__ logits, unsigned* __restrict__ menc, int E){
    int i = blockIdx.x * blockDim.x + threadIdx.x;
    if (i >= 2 * E * NHEAD) return;
    int g = i >> 2, h = i & 3;
    int s, d;
    if (g < E){ s = row[g]; d = col[g]; } else { s = g - E; d = g - E; }
    float l = as_[s * 4 + h] + ad_[d * 4 + h];
    l = l > 0.f ? l : 0.2f * l;             // leaky_relu(0.2)
    logits[i] = l;
    atomicMax(&menc[d * 4 + h], fenc(l));
}

// ================= K8: exp + segment sum =================
__global__ void k_expsum(float* __restrict__ logits, const unsigned* __restrict__ menc,
                         float* __restrict__ ssum, const int* __restrict__ col, int E){
    int i = blockIdx.x * blockDim.x + threadIdx.x;
    if (i >= 2 * E * NHEAD) return;
    int g = i >> 2, h = i & 3;
    int d = (g < E) ? col[g] : g - E;
    float m = fdec(menc[d * 4 + h]);
    float ex = __expf(logits[i] - m);
    logits[i] = ex;
    atomicAdd(&ssum[d * 4 + h], ex);
}

// ================= K9: out = bias_g (float4) =================
__global__ void k_initout(float* __restrict__ out, const float* __restrict__ bias, int E){
    int i = blockIdx.x * blockDim.x + threadIdx.x;
    if (i < E * 64) ((float4*)out)[i] = ((const float4*)bias)[i & 63];
}

// ================= K10: out[dst2] += alpha * xg[src2] (float4 loads) =================
__global__ void k_accum(const float* __restrict__ logits, const float* __restrict__ ssum,
                        const float* __restrict__ xg, const int* __restrict__ row,
                        const int* __restrict__ col, float* __restrict__ out, int E){
    int i = blockIdx.x * blockDim.x + threadIdx.x;
    if (i >= 2 * E * 64) return;
    int g = i >> 6, q = i & 63, h = q >> 4;
    int s, d;
    if (g < E){ s = row[g]; d = col[g]; } else { s = g - E; d = g - E; }
    float alpha = logits[g * 4 + h] / ssum[d * 4 + h];
    float4 v = *(const float4*)(xg + s * 256 + q * 4);
    float* o = out + d * 256 + q * 4;
    atomicAdd(o + 0, alpha * v.x);
    atomicAdd(o + 1, alpha * v.y);
    atomicAdd(o + 2, alpha * v.z);
    atomicAdd(o + 3, alpha * v.w);
}

extern "C" void kernel_launch(void* const* d_in, const int* in_sizes, int n_in,
                              void* d_out, int out_size, void* d_ws, size_t ws_size,
                              hipStream_t stream) {
    const float* x    = (const float*)d_in[0];
    const int*   ei   = (const int*)  d_in[1];
    const float* ea   = (const float*)d_in[2];
    const float* W1   = (const float*)d_in[3];
    const float* b1   = (const float*)d_in[4];
    const float* W2   = (const float*)d_in[5];
    const float* b2   = (const float*)d_in[6];
    const float* We   = (const float*)d_in[7];
    const float* be   = (const float*)d_in[8];
    const float* Wg   = (const float*)d_in[9];
    const float* atts = (const float*)d_in[10];
    const float* attd = (const float*)d_in[11];
    const float* biasg= (const float*)d_in[12];

    int E = in_sizes[1] / 2;        // edge_index is [2,E]
    int N = in_sizes[0] / NDIM;
    int Epad = (E + 63) & ~63;      // xg padded so GEMM stores are unconditional
    const int* row = ei;
    const int* col = ei + E;

    char* ws = (char*)d_ws;
    size_t off = 0;
    auto alloc = [&](size_t bytes) -> void* {
        void* p = ws + off;
        off = (off + bytes + 255) & ~(size_t)255;
        return p;
    };
    float*          xg   = (float*)         alloc((size_t)Epad * 256 * 4);
    float*          as_  = (float*)         alloc((size_t)E * 4 * 4);
    float*          ad_  = (float*)         alloc((size_t)E * 4 * 4);
    float*          lgt  = (float*)         alloc((size_t)2 * E * 4 * 4);
    unsigned*       menc = (unsigned*)      alloc((size_t)E * 4 * 4);
    float*          ssum = (float*)         alloc((size_t)E * 4 * 4);
    unsigned*       xne  = (unsigned*)      alloc((size_t)N * 64 * 4);
    unsigned short* xnb  = (unsigned short*)alloc((size_t)N * 64 * 2);
    unsigned short* wgs  = (unsigned short*)alloc((size_t)32768 * 2);
    unsigned short* w2s  = (unsigned short*)alloc((size_t)4096 * 2);
    (void)ws_size; (void)n_in; (void)out_size;

    int n1 = N * 64, n2 = E * NHEAD;
    int nmax = n1 > n2 ? n1 : n2;

    k_init    <<<(nmax + 255) / 256, 256, 0, stream>>>(xne, n1, menc, n2, ssum, n2);
    k_w2swz   <<<(4096 + 255) / 256, 256, 0, stream>>>(W2, w2s);
    k_wgswz   <<<(32768 + 255) / 256, 256, 0, stream>>>(Wg, wgs);
    k_edgeconv<<<(E + 63) / 64, 128, 0, stream>>>(x, row, col, W1, b1, w2s, b2, xne, E);
    k_decode  <<<(n1 + 255) / 256, 256, 0, stream>>>(xne, xnb, n1);
    k_gat_gemm<<<(E + 63) / 64, 128, 0, stream>>>(xnb, row, ea, We, be, wgs, xg, E);
    k_att     <<<(E * NHEAD + 255) / 256, 256, 0, stream>>>(xg, atts, attd, as_, ad_, E);
    k_logits  <<<(2 * E * NHEAD + 255) / 256, 256, 0, stream>>>(as_, ad_, row, col, lgt, menc, E);
    k_expsum  <<<(2 * E * NHEAD + 255) / 256, 256, 0, stream>>>(lgt, menc, ssum, col, E);
    k_initout <<<(E * 64 + 255) / 256, 256, 0, stream>>>((float*)d_out, biasg, E);
    k_accum   <<<((unsigned)(2 * E) * 64 + 255) / 256, 256, 0, stream>>>(lgt, ssum, xg, row, col, (float*)d_out, E);
}
